// SelfAttentionModule_53463752900888
// MI455X (gfx1250) — matrix operand order repaired and checked by reading, not compiled, not run on hardware
//
#include <hip/hip_runtime.h>
#include <hip/hip_bf16.h>

// ---------------------------------------------------------------------------
// CDNA5 (gfx1250) fused self-attention: bf16 WMMA, f32 accum,
// double-buffered async global->LDS staging of K/V tiles (ASYNCcnt).
// ---------------------------------------------------------------------------

typedef __attribute__((ext_vector_type(16))) __bf16 v16bf;
typedef __attribute__((ext_vector_type(8)))  float  v8f;

union V8F { v8f v; float f[8]; };

__device__ __forceinline__ unsigned short f2bf(float f) {
  unsigned int u = __float_as_uint(f);
  unsigned int r = 0x7FFFu + ((u >> 16) & 1u);   // round-to-nearest-even
  return (unsigned short)((u + r) >> 16);
}

__device__ __forceinline__ v8f wmma_bf16x32(const unsigned int a[8],
                                            const unsigned int b[8], v8f c) {
  union { unsigned int u[8]; v16bf v; } A, B;
#pragma unroll
  for (int i = 0; i < 8; ++i) { A.u[i] = a[i]; B.u[i] = b[i]; }
  // (neg_a, A, neg_b, B, c_mod, C, reuse_a, reuse_b)
  return __builtin_amdgcn_wmma_f32_16x16x32_bf16(false, A.v, false, B.v,
                                                 (short)0, c, false, false);
}

// Async copy of 16 bytes global -> LDS, tracked with ASYNCcnt.
// Generic LDS pointer: addr[31:0] is the wave-relative LDS byte address (ISA 10.2).
__device__ __forceinline__ void async_load_b128(void* lds_dst, const void* gsrc) {
  unsigned lds = (unsigned)(uintptr_t)lds_dst;
  unsigned long long ga = (unsigned long long)(uintptr_t)gsrc;
  asm volatile("global_load_async_to_lds_b128 %0, %1, off"
               :: "v"(lds), "v"(ga) : "memory");
}
__device__ __forceinline__ void wait_async0() {
  asm volatile("s_wait_asynccnt 0" ::: "memory");
}

// A-tile (16 rows x 32 K) from row-major bf16 src, leading dim ld (elements).
// ISA layout: lane = M (mod 16); VGPR j -> K = (j<4 ? 2j : 16+2(j-4)) + 8*(lane>=16)
__device__ __forceinline__ void load_a_tile(unsigned int a[8], const unsigned short* src,
                                            int row0, int k0, int ld, int lane) {
  const int row = row0 + (lane & 15);
  const int hi8 = (lane >> 4) * 8;
  const unsigned short* p = src + (size_t)row * ld + k0;
#pragma unroll
  for (int j = 0; j < 8; ++j) {
    const int kb = ((j < 4) ? (2 * j) : (16 + 2 * (j - 4))) + hi8;
    a[j] = *(const unsigned int*)(p + kb);       // contiguous bf16 pair
  }
}

// B-tile (32 K x 16 N) where the source matrix Mrm is stored [N-dim][K-dim]
// row-major (ld = K stride): WMMA-B column n = Mrm row, K pairs contiguous.
// ISA layout: lane = N (mod 16); VGPR j -> K = (j<4 ? 2j : 16+2(j-4)) + 8*(lane>=16)
__device__ __forceinline__ void load_bt_tile(unsigned int b[8], const unsigned short* Mrm,
                                             int n0, int d0, int ld, int lane) {
  const int nrow = n0 + (lane & 15);
  const int dbase = d0 + (lane >> 4) * 8;
  const unsigned short* p = Mrm + (size_t)nrow * ld + dbase;
#pragma unroll
  for (int j = 0; j < 8; ++j)
    b[j] = *(const unsigned int*)(p + ((j < 4) ? (2 * j) : (16 + 2 * (j - 4))));
}

// ---------------------------------------------------------------------------
__global__ void sa_cvt_bf16(const float* __restrict__ src,
                            unsigned short* __restrict__ dst, int n) {
  int i = blockIdx.x * blockDim.x + threadIdx.x;
  if (i < n) dst[i] = f2bf(src[i]);
}

// src f32 [K][N] row-major -> dst bf16 [N][K] (transposed, K contiguous)
__global__ void sa_cvt_bf16_T(const float* __restrict__ src,
                              unsigned short* __restrict__ dst, int K, int N) {
  int i = blockIdx.x * blockDim.x + threadIdx.x;
  if (i < K * N) {
    int k = i / N, n = i % N;
    dst[(size_t)n * K + k] = f2bf(src[i]);
  }
}

// Y = X[M x K](bf16) @ W (W given transposed: Wt[Nout x K] bf16) + bias.
// storeT==0: Y row-major [M x Nout] bf16.  storeT==1: Y transposed [Nout x ldT].
__global__ void sa_gemm_proj(const unsigned short* __restrict__ X,
                             const unsigned short* __restrict__ Wt,
                             const float* __restrict__ bias,
                             unsigned short* __restrict__ Y,
                             int M, int K, int Nout, int storeT, int ldT) {
  const int lane = threadIdx.x & 31;
  const int wave = threadIdx.x >> 5;
  const int gw = blockIdx.x * (blockDim.x >> 5) + wave;
  const int ntiles = Nout >> 4;
  const int total = (M >> 4) * ntiles;
  if (gw >= total) return;
  const int mt = gw / ntiles, n0 = (gw % ntiles) << 4;

  V8F acc;
#pragma unroll
  for (int j = 0; j < 8; ++j) acc.f[j] = 0.f;

  unsigned int a[8], b[8];
  for (int k0 = 0; k0 < K; k0 += 32) {
    load_a_tile(a, X, mt << 4, k0, K, lane);
    load_bt_tile(b, Wt, n0, k0, K, lane);
    acc.v = wmma_bf16x32(a, b, acc.v);
  }
  const int hi = lane >> 4, col = lane & 15;
  const float bv = bias[n0 + col];
#pragma unroll
  for (int j = 0; j < 8; ++j) {
    const int row = (mt << 4) + j + hi * 8;
    const unsigned short h = f2bf(acc.f[j] + bv);
    if (!storeT) Y[(size_t)row * Nout + n0 + col] = h;
    else         Y[(size_t)(n0 + col) * ldT + row] = h;
  }
}

// Flash attention: 1 block = 16 query rows x all 256 channels (8 waves x 32 ch).
// K/V tiles double-buffered via async global->LDS copies (ASYNCcnt); one
// barrier per 32-key iteration; next tile's loads overlap current compute.
__global__ void sa_attn_flash(const unsigned short* __restrict__ Qh,
                              const unsigned short* __restrict__ Kh,
                              const unsigned short* __restrict__ Vt, // [256][Mtot]
                              unsigned short* __restrict__ Oh,
                              int Ntok, int Mld) {
  __shared__ unsigned short Ktile[2][32 * 32];   //  4 KB: [key][32 dims]
  __shared__ unsigned short Vtile[2][256 * 32];  // 32 KB: [channel][32 keys]
  __shared__ unsigned short Plds[8][16 * 32];    //  8 KB: per-wave P transpose

  const int tid  = threadIdx.x;
  const int lane = tid & 31;
  const int wave = tid >> 5;
  const int tilesPerBatch = Ntok >> 4;
  const int b = blockIdx.x / tilesPerBatch;
  const int qrow0 = (blockIdx.x % tilesPerBatch) << 4;

  const unsigned short* Qb = Qh + (size_t)b * Ntok * 32;
  const unsigned short* Kb = Kh + (size_t)b * Ntok * 32;
  const size_t vcolBase = (size_t)b * Ntok;

  // Issue async staging of tile starting at key n0i into buffer buf.
  auto stage = [&](int buf, int n0i) {
    if (tid < 128) {                             // K tile: 128 x 16B
      const unsigned short* g = Kb + (size_t)(n0i + (tid >> 2)) * 32 + (tid & 3) * 8;
      async_load_b128(&Ktile[buf][tid * 8], g);
    }
#pragma unroll
    for (int i = 0; i < 4; ++i) {                // V tile: 1024 x 16B
      const int l16 = i * 256 + tid;
      const int ch = l16 >> 2, chunk = l16 & 3;
      const unsigned short* g = Vt + (size_t)ch * Mld + vcolBase + n0i + chunk * 8;
      async_load_b128(&Vtile[buf][l16 * 8], g);
    }
  };

  unsigned int aQ[8];
  load_a_tile(aQ, Qb, qrow0, 0, 32, lane);       // head dim 32 == one WMMA K

  V8F acc0, acc1;
  float mrow[8], lrow[8];
#pragma unroll
  for (int j = 0; j < 8; ++j) {
    acc0.f[j] = 0.f; acc1.f[j] = 0.f; mrow[j] = -1e30f; lrow[j] = 0.f;
  }
  const int c0 = wave * 32;
  const int hi = lane >> 4, col = lane & 15;
  unsigned short* pl = &Plds[wave][0];

  stage(0, 0);                                   // prologue: fill buffer 0

  for (int n0 = 0; n0 < Ntok; n0 += 32) {
    const int cur = (n0 >> 5) & 1;
    wait_async0();                               // our in-flight tile is done
    __syncthreads();                             // everyone's tile done; prev
                                                 // iter's reads of buf^1 done
    if (n0 + 32 < Ntok) stage(cur ^ 1, n0 + 32); // overlap next tile w/ compute

    // ---- S = Q @ K^T for 32 keys -> two 16x16 f32 C-tiles ----
    unsigned int bK[8];
    V8F s0, s1;
#pragma unroll
    for (int j = 0; j < 8; ++j) { s0.f[j] = 0.f; s1.f[j] = 0.f; }
    load_bt_tile(bK, Ktile[cur], 0, 0, 32, lane);
    s0.v = wmma_bf16x32(aQ, bK, s0.v);
    load_bt_tile(bK, Ktile[cur], 16, 0, 32, lane);
    s1.v = wmma_bf16x32(aQ, bK, s1.v);

    // ---- online softmax (C-layout: VGPR j = row j+8*hi, cols = 16 lanes) ----
#pragma unroll
    for (int j = 0; j < 8; ++j) {
      float t = fmaxf(s0.f[j], s1.f[j]);
#pragma unroll
      for (int o = 1; o < 16; o <<= 1) t = fmaxf(t, __shfl_xor(t, o, 32));
      const float mn = fmaxf(mrow[j], t);
      const float p0 = __expf(s0.f[j] - mn);
      const float p1 = __expf(s1.f[j] - mn);
      float rs = p0 + p1;
#pragma unroll
      for (int o = 1; o < 16; o <<= 1) rs += __shfl_xor(rs, o, 32);
      const float sc = __expf(mrow[j] - mn);
      lrow[j] = lrow[j] * sc + rs;
      mrow[j] = mn;
      acc0.f[j] *= sc; acc1.f[j] *= sc;
      s0.f[j] = p0; s1.f[j] = p1;
    }

    // ---- transpose P: C-layout -> A-layout via private LDS slice ----
#pragma unroll
    for (int j = 0; j < 8; ++j) {
      const int row = j + hi * 8;
      pl[row * 32 + col]      = f2bf(s0.f[j]);
      pl[row * 32 + 16 + col] = f2bf(s1.f[j]);
    }
    asm volatile("s_wait_dscnt 0" ::: "memory");  // intra-wave LDS visibility

    unsigned int aP[8];
    {
      const int arow = lane & 15;
#pragma unroll
      for (int j = 0; j < 8; ++j) {
        const int kb = ((j < 4) ? (2 * j) : (16 + 2 * (j - 4))) + hi * 8;
        aP[j] = *(const unsigned int*)(pl + arow * 32 + kb);
      }
    }

    // ---- O += P @ V (this wave's 32 channels; B from LDS V tile) ----
    unsigned int bV[8];
    load_bt_tile(bV, Vtile[cur], c0, 0, 32, lane);
    acc0.v = wmma_bf16x32(aP, bV, acc0.v);
    load_bt_tile(bV, Vtile[cur], c0 + 16, 0, 32, lane);
    acc1.v = wmma_bf16x32(aP, bV, acc1.v);
  }

  unsigned short* Ob = Oh + (size_t)b * Ntok * 256;
#pragma unroll
  for (int j = 0; j < 8; ++j) {
    const int row = qrow0 + j + hi * 8;
    const float inv = 1.0f / lrow[j];
    Ob[(size_t)row * 256 + c0 + col]      = f2bf(acc0.f[j] * inv);
    Ob[(size_t)row * 256 + c0 + 16 + col] = f2bf(acc1.f[j] * inv);
  }
}

// out = gamma * (Oh @ Wo + bo) + x    (Wo given transposed; M x 256 f32 out)
__global__ void sa_gemm_out(const unsigned short* __restrict__ Oh,
                            const unsigned short* __restrict__ Wot,
                            const float* __restrict__ bo,
                            const float* __restrict__ x,
                            const float* __restrict__ gamma,
                            float* __restrict__ out, int M) {
  const int lane = threadIdx.x & 31;
  const int wave = threadIdx.x >> 5;
  const int gw = blockIdx.x * (blockDim.x >> 5) + wave;
  const int total = (M >> 4) * 16;
  if (gw >= total) return;
  const int mt = gw / 16, n0 = (gw % 16) << 4;

  V8F acc;
#pragma unroll
  for (int j = 0; j < 8; ++j) acc.f[j] = 0.f;

  unsigned int a[8], b[8];
  for (int k0 = 0; k0 < 256; k0 += 32) {
    load_a_tile(a, Oh, mt << 4, k0, 256, lane);
    load_bt_tile(b, Wot, n0, k0, 256, lane);
    acc.v = wmma_bf16x32(a, b, acc.v);
  }
  const int hi = lane >> 4, col = lane & 15;
  const float g = gamma[0];
  const float bv = bo[n0 + col];
#pragma unroll
  for (int j = 0; j < 8; ++j) {
    const size_t idx = (size_t)((mt << 4) + j + hi * 8) * 256 + n0 + col;
    out[idx] = g * (acc.f[j] + bv) + x[idx];
  }
}

// ---------------------------------------------------------------------------
extern "C" void kernel_launch(void* const* d_in, const int* in_sizes, int n_in,
                              void* d_out, int out_size, void* d_ws, size_t ws_size,
                              hipStream_t stream) {
  const float* x     = (const float*)d_in[0];
  const float* Wf    = (const float*)d_in[1];
  const float* bf    = (const float*)d_in[2];
  const float* Wg    = (const float*)d_in[3];
  const float* bg    = (const float*)d_in[4];
  const float* Wh    = (const float*)d_in[5];
  const float* bh    = (const float*)d_in[6];
  const float* Wo    = (const float*)d_in[7];
  const float* bo    = (const float*)d_in[8];
  const float* gamma = (const float*)d_in[9];

  const int Btch = 4, Ntok = 4096, Cc = 256, Kd = 32;
  const int Mtot = Btch * Ntok;                  // 16384 tokens

  char* wsp = (char*)d_ws;
  size_t off = 0;
  auto take = [&](size_t bytes) -> void* {
    void* p = wsp + off;
    off = (off + bytes + 255) & ~(size_t)255;
    return p;
  };
  unsigned short* xh   = (unsigned short*)take((size_t)Mtot * Cc * 2);
  unsigned short* Wft  = (unsigned short*)take((size_t)Cc * Kd * 2);  // [32][256]
  unsigned short* Wgt  = (unsigned short*)take((size_t)Cc * Kd * 2);  // [32][256]
  unsigned short* Wht  = (unsigned short*)take((size_t)Cc * Cc * 2);  // [256][256]
  unsigned short* Wot  = (unsigned short*)take((size_t)Cc * Cc * 2);  // [256][256]
  unsigned short* Qh   = (unsigned short*)take((size_t)Mtot * Kd * 2);
  unsigned short* Kh   = (unsigned short*)take((size_t)Mtot * Kd * 2);
  unsigned short* Vt   = (unsigned short*)take((size_t)Mtot * Cc * 2); // [256][Mtot]
  unsigned short* Oh   = (unsigned short*)take((size_t)Mtot * Cc * 2);

  // 1) fp32 -> bf16 conversions (weights transposed for contiguous B-tiles)
  sa_cvt_bf16 <<<(Mtot * Cc + 255) / 256, 256, 0, stream>>>(x, xh, Mtot * Cc);
  sa_cvt_bf16_T<<<(Cc * Kd + 255) / 256, 256, 0, stream>>>(Wf, Wft, Cc, Kd);
  sa_cvt_bf16_T<<<(Cc * Kd + 255) / 256, 256, 0, stream>>>(Wg, Wgt, Cc, Kd);
  sa_cvt_bf16_T<<<(Cc * Cc + 255) / 256, 256, 0, stream>>>(Wh, Wht, Cc, Cc);
  sa_cvt_bf16_T<<<(Cc * Cc + 255) / 256, 256, 0, stream>>>(Wo, Wot, Cc, Cc);

  // 2) Q, K, V projections (V stored channel-major for attention B-tiles)
  const int tilesQK = (Mtot / 16) * (Kd / 16);   // 2048
  const int tilesV  = (Mtot / 16) * (Cc / 16);   // 16384
  sa_gemm_proj<<<tilesQK / 8, 256, 0, stream>>>(xh, Wft, bf, Qh, Mtot, Cc, Kd, 0, 0);
  sa_gemm_proj<<<tilesQK / 8, 256, 0, stream>>>(xh, Wgt, bg, Kh, Mtot, Cc, Kd, 0, 0);
  sa_gemm_proj<<<tilesV  / 8, 256, 0, stream>>>(xh, Wht, bh, Vt, Mtot, Cc, Cc, 1, Mtot);

  // 3) fused flash attention (1 block per 16-query tile per batch)
  sa_attn_flash<<<Btch * (Ntok / 16), 256, 0, stream>>>(Qh, Kh, Vt, Oh, Ntok, Mtot);

  // 4) output projection + gamma-scaled residual, fp32 out
  sa_gemm_out<<<tilesV / 8, 256, 0, stream>>>(Oh, Wot, bo, x, gamma,
                                              (float*)d_out, Mtot);
}
